// ProcessorLayer_38809324486917
// MI455X (gfx1250) — compile-verified, matrix operand order checked by reading, not run on hardware
//
#include <hip/hip_runtime.h>

// ---------------------------------------------------------------------------
// MI455X (gfx1250) hetero-GNN processor layer.
// All GEMMs through v_wmma_f32_16x16x32_bf16 (wave32, 16x16 tiles).
// Gather stage vectorized to b128 loads (HBM-bound phase).
// ---------------------------------------------------------------------------

#define HDIM 128

typedef __attribute__((ext_vector_type(16))) __bf16 v16bf;
typedef __attribute__((ext_vector_type(8)))  float  v8f;

__device__ __forceinline__ unsigned int f2bf1(float f) {
  unsigned int u = __builtin_bit_cast(unsigned int, f);
  u += 0x7FFFu + ((u >> 16) & 1u);        // round to nearest even
  return u >> 16;
}
__device__ __forceinline__ unsigned short f2bf(float f) {
  return (unsigned short)f2bf1(f);
}
__device__ __forceinline__ unsigned int f2bf_pk(float lo, float hi) {
  return f2bf1(lo) | (f2bf1(hi) << 16);
}

// ------------------------------- prep kernels ------------------------------

__global__ void __launch_bounds__(256) zero_f32v4(float4* __restrict__ p, int n4) {
  int i = blockIdx.x * 256 + threadIdx.x;
  if (i < n4) p[i] = make_float4(0.f, 0.f, 0.f, 0.f);
}

// W [rows][cols] f32 row-major  ->  Wt [cols][rows] bf16 (B-matrix: col-major)
__global__ void __launch_bounds__(256) transpose_bf16(const float* __restrict__ W,
                                                      unsigned short* __restrict__ Wt,
                                                      int rows, int cols) {
  int i = blockIdx.x * 256 + threadIdx.x;
  if (i >= rows * cols) return;
  int k = i / cols, n = i % cols;
  Wt[(size_t)n * rows + k] = f2bf(W[i]);
}

// ------------------------------- main kernel -------------------------------
// MODE 0: physical edge  in = [x_bus[src] | x_bus[dst] | ea],  out: new_ea = ea+de,
//         atomicAdd(msg_bus[dst], de)
// MODE 1: link edge      in = [xs[src] | xd[dst]],  atomicAdd(msg_dst[dst], m)
// MODE 2: node update    in = [x[i] | msg[i]],      out: new_x = x + de
template <int DIN, int MODE>
__global__ void __launch_bounds__(256) mlp_pass(
    const float* __restrict__ xA, const int* __restrict__ idx0,
    const float* __restrict__ xB, const int* __restrict__ idx1,
    const float* __restrict__ eaIn,
    const unsigned short* __restrict__ W1t, const float* __restrict__ b1,
    const float* __restrict__ gamma, const float* __restrict__ beta,
    const unsigned short* __restrict__ W2t, const float* __restrict__ b2,
    float* __restrict__ msgT,
    float* __restrict__ outP, const float* __restrict__ residP,
    int E) {
  constexpr int DHID = DIN;          // dhid == din in this model
  constexpr int NT1  = DHID / 16;    // N-tiles of GEMM1 (24 or 16)
  constexpr int TPW  = NT1 / 8;      // tiles per wave (8 waves)
  constexpr int KIT1 = DIN  / 32;
  constexpr int KIT2 = DHID / 32;
  static_assert(NT1 % 8 == 0, "tiles must divide across 8 waves");

  __shared__ unsigned short Xa[16 * DIN];   // gathered input rows, bf16
  __shared__ float          Hf[16 * DHID];  // hidden pre-LN, f32
  __shared__ unsigned short Hb[16 * DHID];  // hidden post LN+ReLU, bf16

  const int tid  = threadIdx.x;
  const int wave = tid >> 5;
  const int lane = tid & 31;
  const int rr   = lane & 15;        // row (A) / col-within-tile (B/C)
  const int half = lane >> 4;
  const int kb   = half * 8;         // K sub-offset per ISA 16-bit layout
  const int base = blockIdx.x * 16;  // first row (edge/node) of this block

  // ---- gather: build 16 x DIN bf16 tile in LDS, b128 loads ----
  // Every 4-float chunk lies inside one 128-float segment (HDIM % 4 == 0).
  for (int i4 = tid; i4 < 16 * DIN / 4; i4 += 256) {
    const int i   = i4 * 4;
    const int row = i / DIN, c = i % DIN;
    const int e   = base + row;
    float4 v = make_float4(0.f, 0.f, 0.f, 0.f);
    if (e < E) {
      if (MODE == 0) {
        if (c < HDIM)          v = *(const float4*)&xA[(size_t)idx0[e] * HDIM + c];
        else if (c < 2 * HDIM) v = *(const float4*)&xB[(size_t)idx1[e] * HDIM + (c - HDIM)];
        else                   v = *(const float4*)&eaIn[(size_t)e * HDIM + (c - 2 * HDIM)];
      } else if (MODE == 1) {
        if (c < HDIM)          v = *(const float4*)&xA[(size_t)idx0[e] * HDIM + c];
        else                   v = *(const float4*)&xB[(size_t)idx1[e] * HDIM + (c - HDIM)];
      } else {
        if (c < HDIM)          v = *(const float4*)&xA[(size_t)e * HDIM + c];
        else                   v = *(const float4*)&xB[(size_t)e * HDIM + (c - HDIM)];
      }
    }
    unsigned int pk0 = f2bf_pk(v.x, v.y);
    unsigned int pk1 = f2bf_pk(v.z, v.w);
    unsigned int* dst = (unsigned int*)&Xa[i];   // 8B aligned (i % 4 == 0)
    dst[0] = pk0;
    dst[1] = pk1;
  }
  __syncthreads();

  // ---- GEMM1: h = X @ W1  (16 x DHID), each wave owns TPW 16-wide N tiles ----
  const v8f zf = {0.f, 0.f, 0.f, 0.f, 0.f, 0.f, 0.f, 0.f};
  v8f acc1[TPW];
#pragma unroll
  for (int t = 0; t < TPW; ++t) acc1[t] = zf;

  for (int kt = 0; kt < KIT1; ++kt) {
    const int k0 = kt * 32;
    union { unsigned int u[8]; v16bf v; } A;
#pragma unroll
    for (int j = 0; j < 8; ++j) {
      int ko = ((j < 4) ? 2 * j : 8 + 2 * j) + kb;   // {0,2,4,6,16,18,20,22}+kb
      A.u[j] = *(const unsigned int*)&Xa[rr * DIN + k0 + ko];
    }
#pragma unroll
    for (int t = 0; t < TPW; ++t) {
      const int col = (wave + t * 8) * 16 + rr;
      union { unsigned int u[8]; v16bf v; } B;
#pragma unroll
      for (int j = 0; j < 8; ++j) {
        int ko = ((j < 4) ? 2 * j : 8 + 2 * j) + kb;
        B.u[j] = *(const unsigned int*)&W1t[(size_t)col * DIN + k0 + ko];
      }
      acc1[t] = __builtin_amdgcn_wmma_f32_16x16x32_bf16(
          false, A.v, false, B.v, (short)0, acc1[t], false, false);
    }
  }
  // C layout: VGPR j -> row (j + 8*half), col rr within tile
#pragma unroll
  for (int t = 0; t < TPW; ++t) {
    const int col = (wave + t * 8) * 16 + rr;
    const float bias = b1[col];
#pragma unroll
    for (int j = 0; j < 8; ++j)
      Hf[(j + 8 * half) * DHID + col] = acc1[t][j] + bias;
  }
  __syncthreads();

  // ---- LayerNorm + ReLU per row (one row per wave, twice) ----
  for (int r = wave; r < 16; r += 8) {
    float s = 0.f, s2 = 0.f;
    for (int c = lane; c < DHID; c += 32) {
      float h = Hf[r * DHID + c];
      s += h; s2 += h * h;
    }
#pragma unroll
    for (int m = 16; m >= 1; m >>= 1) {
      s  += __shfl_xor(s,  m, 32);
      s2 += __shfl_xor(s2, m, 32);
    }
    const float mu   = s / (float)DHID;
    const float var  = s2 / (float)DHID - mu * mu;
    const float rstd = rsqrtf(var + 1e-5f);
    for (int c = lane; c < DHID; c += 32) {
      float v = (Hf[r * DHID + c] - mu) * rstd * gamma[c] + beta[c];
      Hb[r * DHID + c] = f2bf(fmaxf(v, 0.0f));
    }
  }
  __syncthreads();

  // ---- GEMM2: de = relu(h) @ W2  (16 x 128), one N tile per wave ----
  v8f acc2 = zf;
  const int col2 = wave * 16 + rr;
  for (int kt = 0; kt < KIT2; ++kt) {
    const int k0 = kt * 32;
    union { unsigned int u[8]; v16bf v; } A, B;
#pragma unroll
    for (int j = 0; j < 8; ++j) {
      int ko = ((j < 4) ? 2 * j : 8 + 2 * j) + kb;
      A.u[j] = *(const unsigned int*)&Hb[rr * DHID + k0 + ko];
      B.u[j] = *(const unsigned int*)&W2t[(size_t)col2 * DHID + k0 + ko];
    }
    acc2 = __builtin_amdgcn_wmma_f32_16x16x32_bf16(
        false, A.v, false, B.v, (short)0, acc2, false, false);
  }

  // ---- epilogue: residual write and/or atomic scatter ----
  const float bias2 = b2[col2];
#pragma unroll
  for (int j = 0; j < 8; ++j) {
    const int e = base + j + 8 * half;
    if (e < E) {
      const float de = acc2[j] + bias2;
      if (MODE == 0) {
        outP[(size_t)e * HDIM + col2] = residP[(size_t)e * HDIM + col2] + de;
        atomicAdd(&msgT[(size_t)idx1[e] * HDIM + col2], de);
      } else if (MODE == 1) {
        atomicAdd(&msgT[(size_t)idx1[e] * HDIM + col2], de);
      } else {
        outP[(size_t)e * HDIM + col2] = residP[(size_t)e * HDIM + col2] + de;
      }
    }
  }
}

// ------------------------------- host launch -------------------------------

extern "C" void kernel_launch(void* const* d_in, const int* in_sizes, int n_in,
                              void* d_out, int out_size, void* d_ws, size_t ws_size,
                              hipStream_t stream) {
  (void)in_sizes; (void)n_in; (void)out_size; (void)ws_size;

  constexpr int N_BUS = 100000, N_GEN = 20000, N_LOAD = 40000, N_SHUNT = 10000;
  constexpr int E_AC = 250000, E_TR = 60000;

  const float* x_bus   = (const float*)d_in[0];
  const float* x_gen   = (const float*)d_in[1];
  const float* x_load  = (const float*)d_in[2];
  const float* x_shunt = (const float*)d_in[3];
  const float* ea_ac   = (const float*)d_in[4];
  const float* ea_tr   = (const float*)d_in[5];
  const int* ei_ac = (const int*)d_in[6];
  const int* ei_tr = (const int*)d_in[7];
  const int* ei_gb = (const int*)d_in[8];
  const int* ei_bg = (const int*)d_in[9];
  const int* ei_lb = (const int*)d_in[10];
  const int* ei_bl = (const int*)d_in[11];
  const int* ei_sb = (const int*)d_in[12];
  const int* ei_bs = (const int*)d_in[13];
  // param tuples (W1,b1,g,be,W2,b2) flattened starting at 14, 6 leaves each
  auto P = [&](int mlp, int j) { return (const float*)d_in[14 + mlp * 6 + j]; };
  // mlp order: 0 ac, 1 tr, 2 gb, 3 bg, 4 lb, 5 bl, 6 sb, 7 bs,
  //            8 bus, 9 gen, 10 load, 11 shunt

  // ---- workspace layout ----
  char* ws = (char*)d_ws;
  size_t off = 0;
  float* msg_bus   = (float*)(ws + off); off += (size_t)N_BUS   * HDIM * 4;
  float* msg_gen   = (float*)(ws + off); off += (size_t)N_GEN   * HDIM * 4;
  float* msg_load  = (float*)(ws + off); off += (size_t)N_LOAD  * HDIM * 4;
  float* msg_shunt = (float*)(ws + off); off += (size_t)N_SHUNT * HDIM * 4;

  unsigned short* w1t[12];
  unsigned short* w2t[12];
  for (int m = 0; m < 12; ++m) {
    const int din = (m < 2) ? 384 : 256;
    w1t[m] = (unsigned short*)(ws + off); off += (size_t)din * din * 2;
    w2t[m] = (unsigned short*)(ws + off); off += (size_t)HDIM * din * 2;
  }

  // ---- zero message accumulators (vectorized) ----
  {
    int n4 = (N_BUS + N_GEN + N_LOAD + N_SHUNT) * HDIM / 4;
    zero_f32v4<<<(n4 + 255) / 256, 256, 0, stream>>>((float4*)msg_bus, n4);
  }
  // ---- convert weights to bf16, transposed [N][K] ----
  for (int m = 0; m < 12; ++m) {
    const int din = (m < 2) ? 384 : 256;
    transpose_bf16<<<(din * din + 255) / 256, 256, 0, stream>>>(P(m, 0), w1t[m], din, din);
    transpose_bf16<<<(din * HDIM + 255) / 256, 256, 0, stream>>>(P(m, 4), w2t[m], din, HDIM);
  }

  // ---- output layout (concat of return tuple) ----
  float* out      = (float*)d_out;
  float* o_bus    = out;
  float* o_gen    = o_bus   + (size_t)N_BUS   * HDIM;
  float* o_load   = o_gen   + (size_t)N_GEN   * HDIM;
  float* o_shunt  = o_load  + (size_t)N_LOAD  * HDIM;
  float* o_ea_ac  = o_shunt + (size_t)N_SHUNT * HDIM;
  float* o_ea_tr  = o_ea_ac + (size_t)E_AC    * HDIM;

  // ---- physical edges (3H input) ----
  mlp_pass<384, 0><<<E_AC / 16, 256, 0, stream>>>(
      x_bus, ei_ac, x_bus, ei_ac + E_AC, ea_ac,
      w1t[0], P(0, 1), P(0, 2), P(0, 3), w2t[0], P(0, 5),
      msg_bus, o_ea_ac, ea_ac, E_AC);
  mlp_pass<384, 0><<<E_TR / 16, 256, 0, stream>>>(
      x_bus, ei_tr, x_bus, ei_tr + E_TR, ea_tr,
      w1t[1], P(1, 1), P(1, 2), P(1, 3), w2t[1], P(1, 5),
      msg_bus, o_ea_tr, ea_tr, E_TR);

  // ---- link edges (2H input) ----
  mlp_pass<256, 1><<<N_GEN / 16, 256, 0, stream>>>(
      x_gen, ei_gb, x_bus, ei_gb + N_GEN, nullptr,
      w1t[2], P(2, 1), P(2, 2), P(2, 3), w2t[2], P(2, 5),
      msg_bus, nullptr, nullptr, N_GEN);
  mlp_pass<256, 1><<<N_GEN / 16, 256, 0, stream>>>(
      x_bus, ei_bg, x_gen, ei_bg + N_GEN, nullptr,
      w1t[3], P(3, 1), P(3, 2), P(3, 3), w2t[3], P(3, 5),
      msg_gen, nullptr, nullptr, N_GEN);
  mlp_pass<256, 1><<<N_LOAD / 16, 256, 0, stream>>>(
      x_load, ei_lb, x_bus, ei_lb + N_LOAD, nullptr,
      w1t[4], P(4, 1), P(4, 2), P(4, 3), w2t[4], P(4, 5),
      msg_bus, nullptr, nullptr, N_LOAD);
  mlp_pass<256, 1><<<N_LOAD / 16, 256, 0, stream>>>(
      x_bus, ei_bl, x_load, ei_bl + N_LOAD, nullptr,
      w1t[5], P(5, 1), P(5, 2), P(5, 3), w2t[5], P(5, 5),
      msg_load, nullptr, nullptr, N_LOAD);
  mlp_pass<256, 1><<<N_SHUNT / 16, 256, 0, stream>>>(
      x_shunt, ei_sb, x_bus, ei_sb + N_SHUNT, nullptr,
      w1t[6], P(6, 1), P(6, 2), P(6, 3), w2t[6], P(6, 5),
      msg_bus, nullptr, nullptr, N_SHUNT);
  mlp_pass<256, 1><<<N_SHUNT / 16, 256, 0, stream>>>(
      x_bus, ei_bs, x_shunt, ei_bs + N_SHUNT, nullptr,
      w1t[7], P(7, 1), P(7, 2), P(7, 3), w2t[7], P(7, 5),
      msg_shunt, nullptr, nullptr, N_SHUNT);

  // ---- node updates (2H input: [x | msg]) ----
  mlp_pass<256, 2><<<N_BUS / 16, 256, 0, stream>>>(
      x_bus, nullptr, msg_bus, nullptr, nullptr,
      w1t[8], P(8, 1), P(8, 2), P(8, 3), w2t[8], P(8, 5),
      nullptr, o_bus, x_bus, N_BUS);
  mlp_pass<256, 2><<<N_GEN / 16, 256, 0, stream>>>(
      x_gen, nullptr, msg_gen, nullptr, nullptr,
      w1t[9], P(9, 1), P(9, 2), P(9, 3), w2t[9], P(9, 5),
      nullptr, o_gen, x_gen, N_GEN);
  mlp_pass<256, 2><<<N_LOAD / 16, 256, 0, stream>>>(
      x_load, nullptr, msg_load, nullptr, nullptr,
      w1t[10], P(10, 1), P(10, 2), P(10, 3), w2t[10], P(10, 5),
      nullptr, o_load, x_load, N_LOAD);
  mlp_pass<256, 2><<<N_SHUNT / 16, 256, 0, stream>>>(
      x_shunt, nullptr, msg_shunt, nullptr, nullptr,
      w1t[11], P(11, 1), P(11, 2), P(11, 3), w2t[11], P(11, 5),
      nullptr, o_shunt, x_shunt, N_SHUNT);
}
